// UnifiedMambaBlock_91018946937020
// MI455X (gfx1250) — matrix operand order repaired and checked by reading, not compiled, not run on hardware
//
#include <hip/hip_runtime.h>
#include <hip/hip_bf16.h>
#include <cstdint>
#include <cstddef>

// ---------------------------------------------------------------------------
// Mamba block forward for MI455X (gfx1250, wave32, WMMA + TDM).
// GEMMs: v_wmma_f32_16x16x32_bf16; B panels staged to LDS by the Tensor Data
// Mover (tensor_load_to_lds, TENSORcnt double-buffered), A streamed from
// global with prefetch. Operands pre-packed into WMMA fragments with
// contiguous k = 16*(lane/16) + i per lane. Selective scan runs as a 3-pass chunked
// parallel scan (64 chunks x 64 steps) for full-device occupancy.
// ---------------------------------------------------------------------------

#define SEQ_LEN  4096
#define DMODEL   768
#define DINNER   1536
#define DTRANK   48
#define DSTATE   16
#define NCHUNK   64
#define LCHUNK   (SEQ_LEN / NCHUNK)   // 64

typedef __attribute__((ext_vector_type(16))) __bf16 v16bf;
typedef __attribute__((ext_vector_type(8)))  float  v8f;
typedef __attribute__((ext_vector_type(4)))  unsigned int u32x4;
typedef __attribute__((ext_vector_type(8)))  int          i32x8;
typedef __attribute__((ext_vector_type(4)))  int          i32x4;

#if __has_builtin(__builtin_amdgcn_tensor_load_to_lds) && \
    __has_builtin(__builtin_amdgcn_s_wait_tensorcnt)
#define USE_TDM 1
#else
#define USE_TDM 0
#endif

__device__ __forceinline__ unsigned short f2bf(float f) {
  // round-to-nearest-even f32 -> bf16
  unsigned int u = __float_as_uint(f);
  unsigned int r = u + 0x7FFFu + ((u >> 16) & 1u);
  return (unsigned short)(r >> 16);
}

// ---------------------------------------------------------------------------
// Pack A (row-major f32 [M=4096, ld], logical K = Klog, zero-pad to 32*numTilesK)
// into WMMA A-fragment order: [tileM][tileK][lane][16 bf16].
// ---------------------------------------------------------------------------
__global__ __launch_bounds__(256) void k_pack_a(const float* __restrict__ src,
                                                unsigned short* __restrict__ dst,
                                                int ld, int Klog, int numTilesK) {
  size_t tid = (size_t)blockIdx.x * 256 + threadIdx.x;
  int i    = (int)(tid & 15);
  int lane = (int)((tid >> 4) & 31);
  size_t t2 = tid >> 9;
  int tileK = (int)(t2 % (size_t)numTilesK);
  int tileM = (int)(t2 / (size_t)numTilesK);
  int half = lane >> 4;
  int m  = tileM * 16 + (lane & 15);
  int kl = i + 16 * half;
  int k  = tileK * 32 + kl;
  float v = (k < Klog) ? src[(size_t)m * ld + k] : 0.0f;
  dst[tid] = f2bf(v);
}

// ---------------------------------------------------------------------------
// Pack B from weight W (row-major f32 [N, ldw]): B = W^T.
// Fragment order: [tileN][tileK][lane][16 bf16]; for one tileN the whole K
// panel is contiguous (numTilesK*1024 bytes) -> TDM-friendly.
// ---------------------------------------------------------------------------
__global__ __launch_bounds__(256) void k_pack_b(const float* __restrict__ W,
                                                unsigned short* __restrict__ dst,
                                                int ldw, int Klog, int numTilesK,
                                                int tileNOff) {
  size_t tid = (size_t)blockIdx.x * 256 + threadIdx.x;
  int i    = (int)(tid & 15);
  int lane = (int)((tid >> 4) & 31);
  size_t t2 = tid >> 9;
  int tileK  = (int)(t2 % (size_t)numTilesK);
  int tileNl = (int)(t2 / (size_t)numTilesK);
  int n  = tileNl * 16 + (lane & 15);
  int kl = i + 16 * (lane >> 4);
  int k  = tileK * 32 + kl;
  float v = (k < Klog) ? W[(size_t)n * ldw + k] : 0.0f;
  size_t di = ((((size_t)(tileNOff + tileNl) * numTilesK) + tileK) * 32 + lane) * 16 + i;
  dst[di] = f2bf(v);
}

// ---------------------------------------------------------------------------
// WMMA GEMM, TDM-fed: C[4096, N] f32 = A(packed bf16) x B(packed bf16).
// 8 waves/block; wave 0 drives the Tensor Data Mover (one 2D D# per chunk,
// double-buffered in LDS, TENSORcnt-tracked); waves consume via ds_load_b128.
// ---------------------------------------------------------------------------
template <int NT>
__global__ __launch_bounds__(256) void k_wmma_gemm(const unsigned short* __restrict__ Asw,
                                                   const unsigned short* __restrict__ Bsw,
                                                   float* __restrict__ C,
                                                   int numTilesK, int ldc) {
  constexpr int CK = 2;                      // k-tiles per TDM chunk
  constexpr int CHUNK_USH = NT * CK * 512;   // ushorts per LDS buffer
  const int lane   = threadIdx.x & 31;
  const int wave   = threadIdx.x >> 5;
  const int tileM  = blockIdx.y * 8 + wave;
  const int nTile0 = blockIdx.x * NT;

  const unsigned short* aPtr = Asw + (((size_t)tileM * numTilesK) * 32 + lane) * 16;

  v8f zero = {};
  v8f acc[NT];
#pragma unroll
  for (int j = 0; j < NT; ++j) acc[j] = zero;

#if USE_TDM
  __shared__ __align__(16) unsigned short shB[2][CHUNK_USH];
  const int nChunks = numTilesK / CK;

  const unsigned int td0 = (unsigned int)(CK * 256);      // chunk words (dim0)
  const unsigned int st0 = (unsigned int)(numTilesK * 256);
  i32x8 g1 = { (int)(2u << 16),                 // workgroup_mask=0, data_size=4B
               (int)(td0 << 16),                // abar=0 | tensor_dim0[15:0]
               (int)((td0 >> 16) | ((unsigned)NT << 16)),  // td0 hi | tensor_dim1 lo
               (int)(((unsigned)NT >> 16) | (td0 << 16)),  // td1 hi | tile_dim0
               (int)NT,                         // tile_dim1 | tile_dim2=0
               (int)st0,                        // tensor_dim0_stride[31:0]
               (int)(st0 >> 16 >> 16),          // stride hi16 | td1_stride lo16 (=0)
               0 };
  i32x4 gz4 = { 0, 0, 0, 0 };
  i32x8 gz8 = { 0, 0, 0, 0, 0, 0, 0, 0 };

  auto tdm_issue = [&](int chunk, int buf) {
    unsigned long long gb = (unsigned long long)(uintptr_t)Bsw +
        ((size_t)nTile0 * numTilesK + (size_t)chunk * CK) * 1024u;
    unsigned int ldsa = (unsigned int)(uintptr_t)(&shB[buf][0]);
    u32x4 g0 = { 1u,                            // count=1, user D#
                 ldsa,                          // lds_addr
                 (unsigned int)gb,              // global_addr[31:0]
                 (unsigned int)(gb >> 32) | (2u << 30) };  // addr[56:32] | type=2
    __builtin_amdgcn_tensor_load_to_lds(g0, g1, gz4, gz4, gz8, 0);
  };

  if (wave == 0) tdm_issue(0, 0);

  for (int c = 0; c < nChunks; ++c) {
    const int buf = c & 1;
    if (wave == 0) {
      if (c + 1 < nChunks) {
        tdm_issue(c + 1, buf ^ 1);
        __builtin_amdgcn_s_wait_tensorcnt(1);   // chunk c landed (in-order)
      } else {
        __builtin_amdgcn_s_wait_tensorcnt(0);
      }
    }
    __syncthreads();                            // publish LDS buffer
#pragma unroll
    for (int ktl = 0; ktl < CK; ++ktl) {
      v16bf a = *reinterpret_cast<const v16bf*>(aPtr);
      aPtr += 512;
      __builtin_prefetch(aPtr, 0, 1);           // global_prefetch_b8 (A stream)
#pragma unroll
      for (int j = 0; j < NT; ++j) {
        const v16bf b = *reinterpret_cast<const v16bf*>(
            &shB[buf][(size_t)(j * CK + ktl) * 512 + lane * 16]);
        acc[j] = __builtin_amdgcn_wmma_f32_16x16x32_bf16(
            false, a, false, b, (short)0, acc[j], false, false);
      }
    }
    __syncthreads();                            // buffer reusable for chunk c+2
  }
#else
  for (int kt = 0; kt < numTilesK; ++kt) {
    v16bf a = *reinterpret_cast<const v16bf*>(aPtr);
    aPtr += 512;
    __builtin_prefetch(aPtr, 0, 1);
#pragma unroll
    for (int j = 0; j < NT; ++j) {
      const unsigned short* bPtr =
          Bsw + ((((size_t)(nTile0 + j) * numTilesK) + kt) * 32 + lane) * 16;
      v16bf b = *reinterpret_cast<const v16bf*>(bPtr);
      acc[j] = __builtin_amdgcn_wmma_f32_16x16x32_bf16(
          false, a, false, b, (short)0, acc[j], false, false);
    }
  }
#endif

  // C/D 16x16 f32 layout: VGPR r -> M = r + 8*(lane>>4), N = lane&15
  const int row0 = tileM * 16 + (lane >> 4) * 8;
  const int col  = lane & 15;
#pragma unroll
  for (int j = 0; j < NT; ++j)
#pragma unroll
    for (int r = 0; r < 8; ++r)
      C[(size_t)(row0 + r) * ldc + (size_t)(nTile0 + j) * 16 + col] = acc[j][r];
}

// ---------------------------------------------------------------------------
// Causal depthwise conv1d (width 4) + bias + SiLU.
// ---------------------------------------------------------------------------
__global__ __launch_bounds__(256) void k_conv_silu(const float* __restrict__ xz,
                                                   const float* __restrict__ cw,
                                                   const float* __restrict__ cb,
                                                   float* __restrict__ xc) {
  int d = blockIdx.x * 256 + threadIdx.x;  // 0..1535
  int t = blockIdx.y;                      // 0..4095
  float acc = cb[d];
#pragma unroll
  for (int j = 0; j < 4; ++j) {
    int tt = t - 3 + j;
    if (tt >= 0) acc += cw[d * 4 + j] * xz[(size_t)tt * (2 * DINNER) + d];
  }
  float s = acc / (1.0f + __expf(-acc));   // silu
  xc[(size_t)t * DINNER + d] = s;
}

// dt = softplus(dt_pre + dt_b), in place
__global__ __launch_bounds__(256) void k_softplus(float* __restrict__ dt,
                                                  const float* __restrict__ dtb) {
  size_t i = (size_t)blockIdx.x * 256 + threadIdx.x;
  int d = (int)(i % DINNER);
  float x = dt[i] + dtb[d];
  dt[i] = (x > 20.0f) ? x : log1pf(__expf(x));
}

// ---------------------------------------------------------------------------
// Chunked parallel selective scan (h_t = dA_t*h_{t-1} + x_t*dt_t*B_t).
// Pass 1: per (chunk, channel) compute chunk decay product aP[s] and local
//         end state hE[s] starting from h=0.   Grid (NCHUNK, 6).
// Pass 2: per (d,s) sequential combine over chunks -> chunk START states
//         in place in chH.                      Grid 96.
// Pass 3: re-scan each chunk from its start state; emit y*silu(z) into the
//         dead x-half of xz.                    Grid (NCHUNK, 6).
// ---------------------------------------------------------------------------
__global__ __launch_bounds__(256) void k_scan_pass1(const float* __restrict__ dt,
                                                    const float* __restrict__ xc,
                                                    const float* __restrict__ P,
                                                    const float* __restrict__ alog,
                                                    float* __restrict__ chA,
                                                    float* __restrict__ chH) {
  int d = blockIdx.y * 256 + threadIdx.x;
  int c = blockIdx.x;
  float A[DSTATE], aP[DSTATE], h[DSTATE];
#pragma unroll
  for (int s = 0; s < DSTATE; ++s) {
    A[s]  = -__expf(alog[(size_t)d * DSTATE + s]);
    aP[s] = 1.0f;
    h[s]  = 0.0f;
  }
  int t0 = c * LCHUNK;
  for (int t = t0; t < t0 + LCHUNK; ++t) {
    float dtv = dt[(size_t)t * DINNER + d];
    float xv  = xc[(size_t)t * DINNER + d];
    float xdt = xv * dtv;
    const float* Bt = P + (size_t)t * 80 + 48;
#pragma unroll
    for (int s = 0; s < DSTATE; ++s) {
      float dA = __expf(dtv * A[s]);
      h[s]  = h[s] * dA + xdt * Bt[s];
      aP[s] *= dA;
    }
  }
  size_t base = ((size_t)c * DINNER + d) * DSTATE;
#pragma unroll
  for (int s = 0; s < DSTATE; ++s) {
    chA[base + s] = aP[s];
    chH[base + s] = h[s];
  }
}

__global__ __launch_bounds__(256) void k_scan_combine(const float* __restrict__ chA,
                                                      float* __restrict__ chH) {
  int i = blockIdx.x * 256 + threadIdx.x;   // flattened (d,s): 24576
  float h = 0.0f;
  for (int c = 0; c < NCHUNK; ++c) {
    size_t idx = (size_t)c * (DINNER * DSTATE) + i;
    float hl = chH[idx];
    float a  = chA[idx];
    chH[idx] = h;            // start state for chunk c
    h = a * h + hl;
  }
}

__global__ __launch_bounds__(256) void k_scan_pass2(const float* __restrict__ dt,
                                                    const float* __restrict__ xc,
                                                    const float* __restrict__ P,
                                                    const float* __restrict__ alog,
                                                    const float* __restrict__ Dp,
                                                    const float* __restrict__ chH,
                                                    float* __restrict__ xz) {
  int d = blockIdx.y * 256 + threadIdx.x;
  int c = blockIdx.x;
  float A[DSTATE], h[DSTATE];
  size_t base = ((size_t)c * DINNER + d) * DSTATE;
#pragma unroll
  for (int s = 0; s < DSTATE; ++s) {
    A[s] = -__expf(alog[(size_t)d * DSTATE + s]);
    h[s] = chH[base + s];
  }
  float Dv = Dp[d];
  int t0 = c * LCHUNK;
  for (int t = t0; t < t0 + LCHUNK; ++t) {
    float dtv = dt[(size_t)t * DINNER + d];
    float xv  = xc[(size_t)t * DINNER + d];
    const float* Bt = P + (size_t)t * 80 + 48;
    const float* Ct = P + (size_t)t * 80 + 64;
    float y = Dv * xv;
    float xdt = xv * dtv;
#pragma unroll
    for (int s = 0; s < DSTATE; ++s) {
      float dA = __expf(dtv * A[s]);
      h[s] = h[s] * dA + xdt * Bt[s];
      y += h[s] * Ct[s];
    }
    float zv = xz[(size_t)t * (2 * DINNER) + DINNER + d];
    y *= zv / (1.0f + __expf(-zv));          // * silu(z)
    xz[(size_t)t * (2 * DINNER) + d] = y;    // y into dead x-half
  }
}

// ---------------------------------------------------------------------------
extern "C" void kernel_launch(void* const* d_in, const int* in_sizes, int n_in,
                              void* d_out, int out_size, void* d_ws, size_t ws_size,
                              hipStream_t stream) {
  (void)in_sizes; (void)n_in; (void)out_size; (void)ws_size;

  const float* u    = (const float*)d_in[0];
  const float* w_in = (const float*)d_in[1];
  const float* cw   = (const float*)d_in[2];
  const float* cb   = (const float*)d_in[3];
  const float* xdtw = (const float*)d_in[4];
  const float* xbw  = (const float*)d_in[5];
  const float* xcw  = (const float*)d_in[6];
  const float* dtw  = (const float*)d_in[7];
  const float* dtb  = (const float*)d_in[8];
  const float* alog = (const float*)d_in[9];
  const float* Dp   = (const float*)d_in[10];
  const float* wout = (const float*)d_in[11];
  float* out = (float*)d_out;

  char* ws = (char*)d_ws;
  size_t off = 0;
  auto alloc = [&](size_t bytes) {
    size_t o = off;
    off += (bytes + 255) & ~(size_t)255;
    return o;
  };
  float*          xz      = (float*)(ws + alloc((size_t)SEQ_LEN * 2 * DINNER * 4));  // xz, later y|z
  float*          xconv   = (float*)(ws + alloc((size_t)SEQ_LEN * DINNER * 4));
  float*          P       = (float*)(ws + alloc((size_t)SEQ_LEN * 80 * 4));          // dt_low|B|C
  float*          dtbuf   = (float*)(ws + alloc((size_t)SEQ_LEN * DINNER * 4));
  float*          chA     = (float*)(ws + alloc((size_t)NCHUNK * DINNER * DSTATE * 4));
  float*          chH     = (float*)(ws + alloc((size_t)NCHUNK * DINNER * DSTATE * 4));
  unsigned short* usw     = (unsigned short*)(ws + alloc((size_t)SEQ_LEN * DMODEL * 2));
  unsigned short* winsw   = (unsigned short*)(ws + alloc((size_t)(2 * DINNER) * DMODEL * 2));
  unsigned short* xcsw    = (unsigned short*)(ws + alloc((size_t)SEQ_LEN * DINNER * 2));
  unsigned short* wcatsw  = (unsigned short*)(ws + alloc((size_t)80 * DINNER * 2));
  unsigned short* dtlowsw = (unsigned short*)(ws + alloc((size_t)SEQ_LEN * 64 * 2));
  unsigned short* dtwsw   = (unsigned short*)(ws + alloc((size_t)DINNER * 64 * 2));
  unsigned short* ysw     = (unsigned short*)(ws + alloc((size_t)SEQ_LEN * DINNER * 2));
  unsigned short* outwsw  = (unsigned short*)(ws + alloc((size_t)DMODEL * DINNER * 2));

  // ---- 1) xz = u @ in_proj_w.T   (M=4096, N=3072, K=768) -------------------
  k_pack_a<<<dim3(256 * 24 * 2), 256, 0, stream>>>(u, usw, DMODEL, DMODEL, 24);
  k_pack_b<<<dim3(192 * 24 * 2), 256, 0, stream>>>(w_in, winsw, DMODEL, DMODEL, 24, 0);
  k_wmma_gemm<4><<<dim3(48, 32), 256, 0, stream>>>(usw, winsw, xz, 24, 2 * DINNER);

  // ---- 2) x_conv = silu(depthwise_conv(x) + b) -----------------------------
  k_conv_silu<<<dim3(6, SEQ_LEN), 256, 0, stream>>>(xz, cw, cb, xconv);

  // ---- 3) P = x_conv @ [x_dt_w|x_b_w|x_c_w].T  (N=80, K=1536) --------------
  k_pack_a<<<dim3(256 * 48 * 2), 256, 0, stream>>>(xconv, xcsw, DINNER, DINNER, 48);
  k_pack_b<<<dim3(3 * 48 * 2), 256, 0, stream>>>(xdtw, wcatsw, DINNER, DINNER, 48, 0);
  k_pack_b<<<dim3(1 * 48 * 2), 256, 0, stream>>>(xbw,  wcatsw, DINNER, DINNER, 48, 3);
  k_pack_b<<<dim3(1 * 48 * 2), 256, 0, stream>>>(xcw,  wcatsw, DINNER, DINNER, 48, 4);
  k_wmma_gemm<1><<<dim3(5, 32), 256, 0, stream>>>(xcsw, wcatsw, P, 48, 80);

  // ---- 4) dt = softplus(dt_low @ dt_w.T + dt_b)  (N=1536, K=48 pad 64) -----
  k_pack_a<<<dim3(256 * 2 * 2), 256, 0, stream>>>(P, dtlowsw, 80, DTRANK, 2);
  k_pack_b<<<dim3(96 * 2 * 2), 256, 0, stream>>>(dtw, dtwsw, DTRANK, DTRANK, 2, 0);
  k_wmma_gemm<4><<<dim3(24, 32), 256, 0, stream>>>(dtlowsw, dtwsw, dtbuf, 2, DINNER);
  k_softplus<<<dim3((SEQ_LEN * DINNER) / 256), 256, 0, stream>>>(dtbuf, dtb);

  // ---- 5) chunked parallel selective scan ----------------------------------
  k_scan_pass1<<<dim3(NCHUNK, 6), 256, 0, stream>>>(dtbuf, xconv, P, alog, chA, chH);
  k_scan_combine<<<dim3((DINNER * DSTATE) / 256), 256, 0, stream>>>(chA, chH);
  k_scan_pass2<<<dim3(NCHUNK, 6), 256, 0, stream>>>(dtbuf, xconv, P, alog, Dp, chH, xz);

  // ---- 6) out = y @ out_proj_w.T  (M=4096, N=768, K=1536) ------------------
  k_pack_a<<<dim3(256 * 48 * 2), 256, 0, stream>>>(xz, ysw, 2 * DINNER, DINNER, 48);
  k_pack_b<<<dim3(48 * 48 * 2), 256, 0, stream>>>(wout, outwsw, DINNER, DINNER, 48, 0);
  k_wmma_gemm<4><<<dim3(12, 32), 256, 0, stream>>>(ysw, outwsw, out, 48, DMODEL);
}
